// SliceProfileLoss_29652454211809
// MI455X (gfx1250) — matrix operand order repaired and checked by reading, not compiled, not run on hardware
//
#include <hip/hip_runtime.h>
#include <cstdint>
#include <cstddef>

typedef _Float16 v16h __attribute__((ext_vector_type(16)));
typedef float    v8f  __attribute__((ext_vector_type(8)));

#define PI_F     3.14159265358979323846f
#define TWOPI_F  6.28318530717958647692f
#define MASK_EPS 0.001f

// Per-row f16 scales so subnormal-range values (dx ~ 1.2e-5) and large values
// (pd^2*w ~ 1e5) both land in f16 normal range; undone after f32 accumulation.
#define SCALE_PD   (1.0f / 64.0f)
#define SCALE_MASS 65536.0f
#define SCALE_CPU  256.0f

__device__ __forceinline__ float wrap_diff(float d) {
  // python-style mod into [0, 2pi), then shift; dm == -pi -> pi
  float m = fmodf(d + PI_F, TWOPI_F);
  if (m < 0.0f) m += TWOPI_F;
  float dm = m - PI_F;
  if (dm == -PI_F) dm = PI_F;
  return dm;
}

// ---------------- kernel 1a: phase + |xy| ----------------
__global__ void spl_phase_abs(const float* __restrict__ xr, const float* __restrict__ xi,
                              float* __restrict__ phase, float* __restrict__ xyabs, int N) {
  size_t off = (size_t)blockIdx.x * (size_t)N;
  for (int i = threadIdx.x; i < N; i += blockDim.x) {
    float r = xr[off + i], m = xi[off + i];
    phase[off + i] = atan2f(m, r);
    xyabs[off + i] = sqrtf(r * r + m * m);
  }
}

// ---------------- kernel 1b: phase_diff^2 * w  and  unwrap (block scan) ----------------
__global__ void spl_pd_unwrap(const float* __restrict__ phase, const float* __restrict__ posAx,
                              float* __restrict__ Apd, float* __restrict__ pu, int N) {
  int b = blockIdx.x;
  size_t off = (size_t)b * (size_t)N;
  int tid = threadIdx.x;

  // phase_diff = Im( grad(exp(i phi)) / exp(i phi) );  Apd = pd^2 * trapz_weight
  for (int i = tid; i < N; i += 256) {
    float ph0 = phase[off + i];
    float ex = cosf(ph0), ey = sinf(ph0);
    float gx, gy;
    if (i == 0) {
      float ph1 = phase[off + 1];
      float inv = 1.0f / (posAx[1] - posAx[0]);
      gx = (cosf(ph1) - ex) * inv; gy = (sinf(ph1) - ey) * inv;
    } else if (i == N - 1) {
      float phm = phase[off + i - 1];
      float inv = 1.0f / (posAx[N - 1] - posAx[N - 2]);
      gx = (ex - cosf(phm)) * inv; gy = (ey - sinf(phm)) * inv;
    } else {
      float phm = phase[off + i - 1], php = phase[off + i + 1];
      float hl = posAx[i] - posAx[i - 1], hr = posAx[i + 1] - posAx[i];
      float a = -hr / (hl * (hl + hr));
      float bb = (hr - hl) / (hl * hr);
      float cc = hl / (hr * (hl + hr));
      gx = a * cosf(phm) + bb * ex + cc * cosf(php);
      gy = a * sinf(phm) + bb * ey + cc * sinf(php);
    }
    float den = ex * ex + ey * ey;
    float pd = (gy * ex - gx * ey) / den;
    float wn;
    if (i == 0)          wn = 0.5f * (posAx[1] - posAx[0]);
    else if (i == N - 1) wn = 0.5f * (posAx[N - 1] - posAx[N - 2]);
    else                 wn = 0.5f * (posAx[i + 1] - posAx[i - 1]);
    Apd[off + i] = pd * pd * wn;
  }
  __syncthreads();

  // unwrap = inclusive prefix over g[0]=phi[0], g[n]=wrap_diff(phi[n]-phi[n-1])
  __shared__ float part[256];
  int SEG = N / 256;
  int s0 = tid * SEG;
  float sum = 0.0f;
  for (int j = 0; j < SEG; j++) {
    int idx = s0 + j;
    float g = (idx == 0) ? phase[off] : wrap_diff(phase[off + idx] - phase[off + idx - 1]);
    sum += g;
  }
  part[tid] = sum;
  __syncthreads();
  for (int d = 1; d < 256; d <<= 1) {
    float v = (tid >= d) ? part[tid - d] : 0.0f;
    __syncthreads();
    part[tid] += v;
    __syncthreads();
  }
  float run = part[tid] - sum;  // exclusive prefix
  for (int j = 0; j < SEG; j++) {
    int idx = s0 + j;
    float g = (idx == 0) ? phase[off] : wrap_diff(phase[off + idx] - phase[off + idx - 1]);
    run += g;
    pu[off + idx] = run;
  }
}

// ---------------- kernel 1c: pair coefficients for masked trapz of pu ----------------
__global__ void spl_cpu(const float* __restrict__ pu, const float* __restrict__ posAx,
                        float* __restrict__ cpuArr, int N) {
  size_t off = (size_t)blockIdx.x * (size_t)N;
  for (int p = threadIdx.x; p < N; p += blockDim.x) {
    cpuArr[off + p] = (p < N - 1)
        ? 0.5f * (pu[off + p] + pu[off + p + 1]) * (posAx[p + 1] - posAx[p])
        : 0.0f;
  }
}

// ---------------- kernel 2: stream target_xy; WMMA contractions + row sums ----------------
// grid = (N/512, B), block = 256 (8 waves).  Waves 0-3: pointwise-mask chain (pd_int),
// s-tiles 0-3.  Waves 4-7: pair-mask chain (mass + pu_num), s-tiles 0-3.
__global__ void __launch_bounds__(256)
spl_main(const float* __restrict__ tgt, const float* __restrict__ posAx,
         const float* __restrict__ Apd, const float* __restrict__ cpuArr,
         float* __restrict__ rowsum, float* __restrict__ mass,
         float* __restrict__ pdint, float* __restrict__ punum, int N) {
  constexpr int S = 64;
  constexpr int ROWPAD = 68;                 // 4-float pad: kills LDS bank conflicts, keeps 16B align
  constexpr int TILE_BYTES = 32 * ROWPAD * 4;
  __shared__ __align__(16) float Tb[2][32][ROWPAD];
  // Per-chunk A-side inputs, staged once, pre-scaled, pre-converted to f16.
  __shared__ _Float16 ApdS[512];
  __shared__ _Float16 dxS[512];              // index i <-> pair p = c0-1+i
  __shared__ _Float16 cpuS[512];

  const int b = blockIdx.y;
  const int c0 = blockIdx.x * 512;
  const size_t bN = (size_t)b * (size_t)N;
  const float* gB = tgt + bN * S;            // per-batch base, uniform -> SGPR pair
  const uint32_t lbase = (uint32_t)(size_t)(&Tb[0][0][0]);

  const int tid = threadIdx.x;
  const int lane = tid & 31;
  const int w = tid >> 5;
  const int hi = lane >> 4;                  // lane >= 16
  const int m16 = lane & 15;

  // --- stage chunk-local A inputs into LDS (coalesced, once per block) ---
  for (int i = tid; i < 512; i += 256) {
    ApdS[i] = (_Float16)(Apd[bN + c0 + i] * SCALE_PD);
    int p = c0 - 1 + i;
    float dxv = 0.0f, cpv = 0.0f;
    if (p >= 0 && p < N - 1) {
      dxv = (posAx[p + 1] - posAx[p]) * SCALE_MASS;
      cpv = cpuArr[bN + p] * SCALE_CPU;
    }
    dxS[i]  = (_Float16)dxv;
    cpuS[i] = (_Float16)cpv;
  }

  auto issue = [&](int t) {
    int r0 = c0 + t * 32;
    #pragma unroll
    for (int i = 0; i < 2; i++) {
      int c = tid + i * 256;                 // 512 x 16B chunks per 8KB tile
      int row = c >> 4;
      int within = (c & 15) * 16;
      uint32_t la = lbase + (uint32_t)(t & 1) * TILE_BYTES + (uint32_t)row * (ROWPAD * 4) + (uint32_t)within;
      uint32_t go = (uint32_t)r0 * 256u + (uint32_t)c * 16u;   // bytes: r0*S*4 + chunk*16
      asm volatile("global_load_async_to_lds_b128 %0, %1, %2"
                   :: "v"(la), "v"(go), "s"(gB) : "memory");
    }
  };

  issue(0);

  // chunk-boundary mask value (row c0-1) for the pair chain at t==0
  const int scol = (w & 3) * 16 + m16;
  float mprev = 0.0f;
  if (c0 > 0) mprev = gB[(size_t)(c0 - 1) * S + scol];

  v8f acc = {};

  for (int t = 0; t < 16; t++) {
    __syncthreads();                         // everyone done with tile t-1 (frees buf (t+1)&1)
    if (t + 1 < 16) {
      issue(t + 1);
      asm volatile("s_wait_asynccnt 2" ::: "memory");  // own tile-t transfers done (in-order)
    } else {
      asm volatile("s_wait_asynccnt 0" ::: "memory");
    }
    __syncthreads();                         // all waves' tile-t data visible

    const int bi = t & 1, pi = bi ^ 1;
    const int r0 = c0 + t * 32;

    // --- row sums over s (tgt_xy), f32 exact ---
    {
      int row = tid >> 3, prt = tid & 7;
      float s = 0.0f;
      #pragma unroll
      for (int j = 0; j < 8; j++) s += Tb[bi][row][prt * 8 + j];
      s += __shfl_xor(s, 1);
      s += __shfl_xor(s, 2);
      s += __shfl_xor(s, 4);
      if (prt == 0) rowsum[bN + r0 + row] = s;
    }

    // --- WMMA chains ---
    const bool isPair = (w >= 4);
    v16h av, bv;

    if (!isPair) {
      // B[k,s] = mask (f16 0/1).  B layout: half h -> K = h + (hi?16:0)
      #pragma unroll
      for (int h = 0; h < 16; h++) {
        int K = h + (hi ? 16 : 0);
        bv[h] = (Tb[bi][K][scol] > MASK_EPS) ? (_Float16)1.0f : (_Float16)0.0f;
      }
      // A row 0 = Apd (scaled).  A layout: half h -> K = h + (hi?16:0), M = m16
      #pragma unroll
      for (int h = 0; h < 16; h++) {
        _Float16 a = (_Float16)0.0f;
        if (m16 == 0) {
          int K = h + (hi ? 16 : 0);
          a = ApdS[t * 32 + K];
        }
        av[h] = a;
      }
    } else {
      // pairs p = r0-1+K ; B[k,s] = m[p,s]*m[p+1,s].
      // Rows K-1 and K overlap across h: evaluate 17 mask bits, AND adjacent.
      bool bit[17];
      #pragma unroll
      for (int j = 0; j < 17; j++) {
        int row = j - 1 + (hi ? 16 : 0);
        float v;
        if (row >= 0)    v = Tb[bi][row][scol];
        else if (t > 0)  v = Tb[pi][31][scol];
        else             v = mprev;          // c0==0 -> mprev==0 -> no pair
        bit[j] = v > MASK_EPS;
      }
      #pragma unroll
      for (int h = 0; h < 16; h++)
        bv[h] = (bit[h] && bit[h + 1]) ? (_Float16)1.0f : (_Float16)0.0f;

      // A row 0 = dx[p]*S, row 1 = cpu[p]*S  (staged, index = t*32 + K)
      #pragma unroll
      for (int h = 0; h < 16; h++) {
        _Float16 a = (_Float16)0.0f;
        int K = h + (hi ? 16 : 0);
        if (m16 == 0)      a = dxS[t * 32 + K];
        else if (m16 == 1) a = cpuS[t * 32 + K];
        av[h] = a;
      }
    }

    acc = __builtin_amdgcn_wmma_f32_16x16x32_f16(false, av, false, bv, (short)0, acc,
                                                 false, false);
  }

  // D layout: VGPR j, lanes 0-15 -> row j.  Fold rows 0/1 into global accumulators.
  if (hi == 0) {
    int s = (w & 3) * 16 + m16;
    if (w < 4) {
      atomicAdd(&pdint[b * S + s], acc[0] * (1.0f / SCALE_PD));
    } else {
      atomicAdd(&mass[b * S + s],  acc[0] * (1.0f / SCALE_MASS));
      atomicAdd(&punum[b * S + s], acc[1] * (1.0f / SCALE_CPU));
    }
  }
}

// ---------------- kernel 3: finalize the four losses ----------------
__device__ __forceinline__ float blockReduce256(float v, float* red) {
  int tid = threadIdx.x;
  red[tid] = v;
  __syncthreads();
  for (int s = 128; s > 0; s >>= 1) {
    if (tid < s) red[tid] += red[tid + s];
    __syncthreads();
  }
  float r = red[0];
  __syncthreads();
  return r;
}

__global__ void spl_final(const float* __restrict__ z, const float* __restrict__ xyabs,
                          const float* __restrict__ rowsum, const float* __restrict__ posAx,
                          const float* __restrict__ mass, const float* __restrict__ pdint,
                          const float* __restrict__ punum, float* __restrict__ out,
                          int N, int B) {
  constexpr int S = 64;
  __shared__ float red[256];
  __shared__ float mt[64];
  int tid = threadIdx.x;
  float extent = posAx[N - 1] - posAx[0];

  // loss_mxy, loss_mz : trapz as pointwise weights
  float s1 = 0.0f, s2 = 0.0f;
  size_t BN = (size_t)B * (size_t)N;
  for (size_t idx = tid; idx < BN; idx += 256) {
    int n = (int)(idx % (size_t)N);
    float wn;
    if (n == 0)          wn = 0.5f * (posAx[1] - posAx[0]);
    else if (n == N - 1) wn = 0.5f * (posAx[N - 1] - posAx[N - 2]);
    else                 wn = 0.5f * (posAx[n + 1] - posAx[n - 1]);
    float tg = rowsum[idx];
    float d1 = xyabs[idx] - tg;
    s1 += d1 * d1 * wn;
    float tz = sqrtf(1.0f - tg * tg);
    float d2 = z[idx] - tz;
    s2 += d2 * d2 * wn;
  }
  float t1 = blockReduce256(s1, red);
  float t2 = blockReduce256(s2, red);

  // per-batch total mass
  if (tid < B) {
    float m = 0.0f;
    for (int s = 0; s < S; s++) m += mass[tid * S + s];
    mt[tid] = m;
  }
  __syncthreads();

  // loss_pd = mean over (b,s) of pdint/mass_total[b]
  float sp = 0.0f;
  for (int idx = tid; idx < B * S; idx += 256) {
    int b = idx >> 6;
    sp += pdint[idx] / mt[b];
  }
  float t3 = blockReduce256(sp, red);

  // loss_po = mean over (b, S-1) of tan(0.5*(diff(phase_means)-0.5))^2
  float so = 0.0f;
  for (int idx = tid; idx < B * (S - 1); idx += 256) {
    int b = idx / (S - 1);
    int j = idx % (S - 1);
    float pm1 = punum[b * S + j]     / mass[b * S + j];
    float pm2 = punum[b * S + j + 1] / mass[b * S + j + 1];
    float d = (pm2 - pm1) - 0.5f;     // PHASE_OFFSET
    float tn = tanf(0.5f * d);
    so += tn * tn;
  }
  float t4 = blockReduce256(so, red);

  if (tid == 0) {
    out[0] = t1 / ((float)B * extent);
    out[1] = t2 / ((float)B * extent);
    out[2] = t3 / (float)(B * S);
    out[3] = t4 / (float)(B * (S - 1));
  }
}

// ---------------- host-side launch ----------------
extern "C" void kernel_launch(void* const* d_in, const int* in_sizes, int n_in,
                              void* d_out, int out_size, void* d_ws, size_t ws_size,
                              hipStream_t stream) {
  (void)n_in; (void)out_size; (void)ws_size;
  const float* z     = (const float*)d_in[0];
  const float* xr    = (const float*)d_in[1];
  const float* xi    = (const float*)d_in[2];
  const float* posAx = (const float*)d_in[5];
  const float* tgt   = (const float*)d_in[6];

  int N = in_sizes[5];
  int B = in_sizes[0] / N;
  const int S = 64;
  size_t BN = (size_t)B * (size_t)N;

  float* w      = (float*)d_ws;
  float* ph     = w;
  float* xyabs  = w + 1 * BN;
  float* pu     = w + 2 * BN;
  float* Apd    = w + 3 * BN;
  float* cp     = w + 4 * BN;
  float* rowsum = w + 5 * BN;
  float* mass   = w + 6 * BN;
  float* pdint  = mass + (size_t)B * S;
  float* punum  = pdint + (size_t)B * S;

  hipMemsetAsync(mass, 0, 3 * (size_t)B * S * sizeof(float), stream);
  spl_phase_abs<<<B, 256, 0, stream>>>(xr, xi, ph, xyabs, N);
  spl_pd_unwrap<<<B, 256, 0, stream>>>(ph, posAx, Apd, pu, N);
  spl_cpu<<<B, 256, 0, stream>>>(pu, posAx, cp, N);
  spl_main<<<dim3(N / 512, B), 256, 0, stream>>>(tgt, posAx, Apd, cp,
                                                 rowsum, mass, pdint, punum, N);
  spl_final<<<1, 256, 0, stream>>>(z, xyabs, rowsum, posAx, mass, pdint, punum,
                                   (float*)d_out, N, B);
}
